// MambaBackbone_7s_51737176048195
// MI455X (gfx1250) — matrix operand order repaired and checked by reading, not compiled, not run on hardware
//
#include <hip/hip_runtime.h>
#include <hip/hip_bf16.h>

// ---------------------------------------------------------------------------
// MambaBackbone on gfx1250 (MI455X).
// GEMMs -> v_wmma_f32_16x16x32_f16 (f16 in, f32 accum), weight tile staged in
// LDS via Tensor Data Mover (tensor_load_to_lds + s_wait_tensorcnt), read back
// with ds_load_b128. All GEMM shapes are compile-time template params so the
// inner loop is a branch-free clause of loads + wmma.
// Scan / LN / conv / activations -> fp32 VALU.
// ---------------------------------------------------------------------------

#define BQ      4
#define LQ      1024
#define CINQ    30
#define DMODEL  256
#define NLAYERS 4
#define FEATQ   512
#define DSTATE  16
#define DCONV   4
#define DINNER  512
#define DTRANK  16
#define BL      (BQ * LQ)          // 4096 rows
#define EPSQ    1e-5f

typedef __attribute__((ext_vector_type(16))) _Float16     v16h;
typedef __attribute__((ext_vector_type(8)))  _Float16     v8h;
typedef __attribute__((ext_vector_type(8)))  float        v8f;
typedef __attribute__((ext_vector_type(4)))  unsigned int u32x4;
typedef __attribute__((ext_vector_type(8)))  int          i32x8;
typedef __attribute__((ext_vector_type(4)))  int          i32x4;

#if defined(__has_builtin)
#if __has_builtin(__builtin_amdgcn_tensor_load_to_lds) && \
    __has_builtin(__builtin_amdgcn_s_wait_tensorcnt)
#define HAVE_TDM 1
#endif
#endif
#ifndef HAVE_TDM
#define HAVE_TDM 0
#endif

__device__ __forceinline__ float sigmoidf_(float x) { return 1.0f / (1.0f + __expf(-x)); }
__device__ __forceinline__ float siluf_(float x)    { return x * sigmoidf_(x); }
__device__ __forceinline__ float softplusf_(float x){ return (x > 20.0f) ? x : log1pf(__expf(x)); }

// --------------------------- f32 -> f16 convert ----------------------------
__global__ void cvt_f32_f16_kernel(const float* __restrict__ src,
                                   _Float16* __restrict__ dst, int n) {
    int i = blockIdx.x * blockDim.x + threadIdx.x;
    if (i < n) dst[i] = (_Float16)src[i];
}

// --------------------------- input projection ------------------------------
// h[b,l,o] = sum_c inp_w[o,c] * x[b,c,l]   (K=30, tiny: VALU kernel)
__global__ void input_proj_kernel(const float* __restrict__ x,
                                  const float* __restrict__ w,
                                  float* __restrict__ h) {
    int bl = blockIdx.x;              // 0..BL-1
    int o  = threadIdx.x;             // 0..255
    int b  = bl >> 10;
    int l  = bl & (LQ - 1);
    const float* xp = x + (size_t)b * CINQ * LQ + l;
    const float* wr = w + o * CINQ;
    float acc = 0.0f;
#pragma unroll
    for (int c = 0; c < CINQ; ++c) acc += wr[c] * xp[(size_t)c * LQ];
    h[(size_t)bl * DMODEL + o] = acc;
}

// --------------------------- LayerNorm (D=256) -----------------------------
__global__ void layernorm_kernel(const float* __restrict__ h,
                                 const float* __restrict__ g,
                                 const float* __restrict__ beta,
                                 _Float16* __restrict__ out) {
    __shared__ float red[256];
    int row = blockIdx.x;
    int d   = threadIdx.x;
    float v = h[(size_t)row * DMODEL + d];
    red[d] = v; __syncthreads();
    for (int s = 128; s > 0; s >>= 1) { if (d < s) red[d] += red[d + s]; __syncthreads(); }
    float mu = red[0] * (1.0f / DMODEL);
    __syncthreads();
    float c = v - mu;
    red[d] = c * c; __syncthreads();
    for (int s = 128; s > 0; s >>= 1) { if (d < s) red[d] += red[d + s]; __syncthreads(); }
    float var = red[0] * (1.0f / DMODEL);
    float r = rsqrtf(var + EPSQ);
    out[(size_t)row * DMODEL + d] = (_Float16)(c * r * g[d] + beta[d]);
}

// --------------------------- WMMA GEMM (templated shape) -------------------
// Y[M,NN] = A[M,KK](f16) * W[NN,KK]^T(f16)  [+ res], f32 out.
// Block = 4 waves (128 thr): wave w -> rows [blockIdx.x*64 + w*16, +16),
// NSUB N-subtiles of 16 (NSUB*16 <= 64 cols per block).
// B tile (NSUB*16 rows x KK halves <= 64KB) staged in LDS once per block via
// the Tensor Data Mover, then consumed with ds_load_b128.
// A-operand layout (ISA 7.12.2, 16-bit A 16x32): lane m=l&15, khalf=l>>4 holds
// K in [khalf*8,+8) and [16+khalf*8,+8). B-operand: lane n=l&15 holds
// weight row n at K in [khalf*8,+8) and [16+khalf*8,+8) (same as A).
template <int NSUB, int NN, int KK, bool HASRES>
__global__ void __launch_bounds__(128)
gemm_wmma_f16_kernel(const _Float16* __restrict__ A,
                     const _Float16* __restrict__ Bw,
                     const float* __restrict__ res,
                     float* __restrict__ C) {
    constexpr int ROWS = NSUB * 16;        // B rows staged per block
    __shared__ _Float16 sB[ROWS * KK];

    const int wave  = threadIdx.x >> 5;
    const int lane  = threadIdx.x & 31;
    const int n0blk = blockIdx.y * 64;

    // ---- stage B tile (ROWS x KK halves) into LDS ----
#if HAVE_TDM
    if (threadIdx.x < 32) {                 // wave 0 issues one TDM op
        unsigned lds_base = (unsigned)(size_t)(void*)&sB[0]; // flat[31:0] == LDS addr
        unsigned long long ga =
            (unsigned long long)(size_t)(Bw + (size_t)n0blk * KK);
        u32x4 g0;
        g0[0] = 1u;                                         // count=1, user desc
        g0[1] = lds_base;                                   // lds_addr
        g0[2] = (unsigned)(ga & 0xffffffffu);               // global_addr[31:0]
        g0[3] = (unsigned)((ga >> 32) & 0x01ffffffu)        // global_addr[56:32]
                | (2u << 30);                               // type=2 ("image")
        i32x8 g1;
        g1[0] = (int)(1u << 16);                            // data_size=1 (2B)
        g1[1] = (int)(((unsigned)KK & 0xffffu) << 16);      // tensor_dim0[15:0]
        g1[2] = (int)((((unsigned)KK >> 16) & 0xffffu)      // tensor_dim0[31:16]
                | (((unsigned)ROWS & 0xffffu) << 16));      // tensor_dim1[15:0]
        g1[3] = (int)((((unsigned)ROWS >> 16) & 0xffffu)    // tensor_dim1[31:16]
                | ((unsigned)KK << 16));                    // tile_dim0 = KK
        g1[4] = (int)(unsigned)ROWS;                        // tile_dim1=ROWS, tile_dim2=0
        g1[5] = (int)(unsigned)KK;                          // tensor_dim0_stride = KK
        g1[6] = 0;                                          // stride hi / dim1_stride lo
        g1[7] = 0;                                          // dim1_stride hi
        i32x4 g2 = {0, 0, 0, 0};
        i32x4 g3 = {0, 0, 0, 0};
        i32x8 g4 = {0, 0, 0, 0, 0, 0, 0, 0};
        __builtin_amdgcn_tensor_load_to_lds(g0, g1, g2, g3, g4, 0);
        __builtin_amdgcn_s_wait_tensorcnt(0);
    }
    __syncthreads();
#else
    {   // cooperative fallback copy (16B chunks)
        constexpr int KC = KK >> 3;
        for (int c = threadIdx.x; c < ROWS * KC; c += 128) {
            int r = c / KC, kk = (c - r * KC) << 3;
            *(v8h*)&sB[r * KK + kk] = *(const v8h*)&Bw[(size_t)(n0blk + r) * KK + kk];
        }
        __syncthreads();
    }
#endif

    const int mi   = lane & 15;
    const int half = lane >> 4;
    const int rowA = blockIdx.x * 64 + wave * 16 + mi;
    const _Float16* aBase = A + (size_t)rowA * KK + half * 8;
    __builtin_prefetch(aBase, 0, 1);        // global_prefetch of the A stream

    v8f acc[NSUB];
#pragma unroll
    for (int s = 0; s < NSUB; ++s) acc[s] = v8f{};

#pragma unroll 4
    for (int k0 = 0; k0 < KK; k0 += 32) {
        union { v16h v; v8h h[2]; } a;
        a.h[0] = *(const v8h*)(aBase + k0);
        a.h[1] = *(const v8h*)(aBase + k0 + 16);
#pragma unroll
        for (int s = 0; s < NSUB; ++s) {
            union { v16h v; v8h h[2]; } b;
            const _Float16* bp = &sB[(s * 16 + mi) * KK + k0 + half * 8];
            b.h[0] = *(const v8h*)(bp);
            b.h[1] = *(const v8h*)(bp + 16);
            acc[s] = __builtin_amdgcn_wmma_f32_16x16x32_f16(
                         false, a.v, false, b.v, (short)0, acc[s], false, false);
        }
    }

    // C/D layout: VGPR r, lanes 0-15 -> M=r, lanes 16-31 -> M=r+8; N = lane&15.
#pragma unroll
    for (int s = 0; s < NSUB; ++s) {
        int    n    = n0blk + s * 16 + (lane & 15);
        size_t base = ((size_t)(blockIdx.x * 64 + wave * 16 + half * 8)) * NN + n;
#pragma unroll
        for (int r = 0; r < 8; ++r) {
            size_t idx = base + (size_t)r * NN;
            float  v   = acc[s][r];
            if (HASRES) v += res[idx];
            C[idx] = v;
        }
    }
}

// --------------------------- depthwise causal conv + SiLU ------------------
__global__ void conv_silu_kernel(const float* __restrict__ xz,
                                 const float* __restrict__ cw,
                                 const float* __restrict__ cb,
                                 int reverse,
                                 float* __restrict__ xcf,
                                 _Float16* __restrict__ xch) {
    int idx = blockIdx.x * blockDim.x + threadIdx.x;   // BL*DINNER
    int d  = idx & (DINNER - 1);
    int bl = idx >> 9;
    int b  = bl >> 10;
    int ll = bl & (LQ - 1);
    float wv[DCONV];
#pragma unroll
    for (int j = 0; j < DCONV; ++j) wv[j] = cw[d * DCONV + j];
    float acc = cb[d];
#pragma unroll
    for (int j = 0; j < DCONV; ++j) {
        int lm = ll - (DCONV - 1) + j;
        if (lm < 0) continue;
        int t = reverse ? (LQ - 1 - lm) : lm;
        acc += wv[j] * xz[((size_t)(b * LQ + t)) * (2 * DINNER) + d];
    }
    float y = siluf_(acc);
    size_t o = (size_t)bl * DINNER + d;
    xcf[o] = y;
    xch[o] = (_Float16)y;
}

// --------------------------- dt = softplus(dtr @ dt_w^T + dt_b) ------------
__global__ void dt_kernel(const float* __restrict__ proj,
                          const float* __restrict__ dtw,
                          const float* __restrict__ dtb,
                          float* __restrict__ dt) {
    int idx = blockIdx.x * blockDim.x + threadIdx.x;   // BL*DINNER
    int d  = idx & (DINNER - 1);
    int bl = idx >> 9;
    const float* pr = proj + (size_t)bl * 48;
    const float* wr = dtw + d * DTRANK;
    float acc = dtb[d];
#pragma unroll
    for (int r = 0; r < DTRANK; ++r) acc += pr[r] * wr[r];
    dt[(size_t)bl * DINNER + d] = softplusf_(acc);
}

// --------------------------- selective scan --------------------------------
__global__ void __launch_bounds__(128)
scan_kernel(const float* __restrict__ xc,
            const float* __restrict__ dt,
            const float* __restrict__ proj,
            const float* __restrict__ alog,
            const float* __restrict__ dvec,
            float* __restrict__ ys) {
    int idx = blockIdx.x * blockDim.x + threadIdx.x;   // BQ*DINNER = 2048
    int d = idx & (DINNER - 1);
    int b = idx >> 9;
    float A[DSTATE], hst[DSTATE];
#pragma unroll
    for (int n = 0; n < DSTATE; ++n) {
        A[n]   = -__expf(alog[d * DSTATE + n]);
        hst[n] = 0.0f;
    }
    float Dv = dvec[d];
    for (int l = 0; l < LQ; ++l) {
        size_t row = (size_t)(b * LQ + l);
        float u   = xc[row * DINNER + d];
        float dtv = dt[row * DINNER + d];
        const float* pr = proj + row * 48;
        float du = dtv * u;
        float y  = u * Dv;
#pragma unroll
        for (int n = 0; n < DSTATE; ++n) {
            float Bn = pr[DTRANK + n];
            float Cn = pr[DTRANK + DSTATE + n];
            hst[n] = __expf(dtv * A[n]) * hst[n] + du * Bn;
            y += hst[n] * Cn;
        }
        ys[row * DINNER + d] = y;
    }
}

// --------------------------- gate + fwd/bwd combine ------------------------
__global__ void combine_kernel(const float* __restrict__ xz,
                               const float* __restrict__ ysf,
                               const float* __restrict__ ysb,
                               _Float16* __restrict__ yout) {
    int idx = blockIdx.x * blockDim.x + threadIdx.x;   // BL*DINNER
    int d  = idx & (DINNER - 1);
    int bl = idx >> 9;
    int b  = bl >> 10;
    int l  = bl & (LQ - 1);
    float z  = xz[(size_t)bl * (2 * DINNER) + DINNER + d];
    float yf = ysf[(size_t)bl * DINNER + d];
    float yb = ysb[((size_t)(b * LQ + (LQ - 1 - l))) * DINNER + d];
    yout[(size_t)bl * DINNER + d] = (_Float16)(siluf_(z) * (yf + yb));
}

// --------------------------- BatchNorm (batch stats) -----------------------
__global__ void bnstats_kernel(const float* __restrict__ t,
                               float* __restrict__ stats) {
    __shared__ float s1[256], s2[256];
    int f   = blockIdx.x;
    int tid = threadIdx.x;
    float sum = 0.0f, ssq = 0.0f;
    for (int s = tid; s < BL; s += 256) {
        float v = t[(size_t)s * FEATQ + f];
        sum += v; ssq += v * v;
    }
    s1[tid] = sum; s2[tid] = ssq; __syncthreads();
    for (int s = 128; s > 0; s >>= 1) {
        if (tid < s) { s1[tid] += s1[tid + s]; s2[tid] += s2[tid + s]; }
        __syncthreads();
    }
    if (tid == 0) {
        float mu  = s1[0] * (1.0f / BL);
        float var = s2[0] * (1.0f / BL) - mu * mu;
        stats[f]         = mu;
        stats[FEATQ + f] = rsqrtf(var + EPSQ);
    }
}

__global__ void bnapply_kernel(const float* __restrict__ t,
                               const float* __restrict__ stats,
                               const float* __restrict__ g,
                               const float* __restrict__ be,
                               float* __restrict__ out) {
    int idx = blockIdx.x * blockDim.x + threadIdx.x;   // BL*FEAT
    int f  = idx & (FEATQ - 1);
    int bl = idx >> 9;
    int b  = bl >> 10;
    int l  = bl & (LQ - 1);
    float v = t[(size_t)bl * FEATQ + f];
    float y = (v - stats[f]) * stats[FEATQ + f] * g[f] + be[f];
    out[((size_t)b * FEATQ + f) * LQ + l] = y;
}

// ---------------------------------------------------------------------------
extern "C" void kernel_launch(void* const* d_in, const int* in_sizes, int n_in,
                              void* d_out, int out_size, void* d_ws, size_t ws_size,
                              hipStream_t stream) {
    (void)in_sizes; (void)n_in; (void)out_size; (void)ws_size;
    const float* x      = (const float*)d_in[0];
    const float* inp_w  = (const float*)d_in[1];
    const float* ln_g   = (const float*)d_in[2];
    const float* ln_b   = (const float*)d_in[3];
    const float* in_w   = (const float*)d_in[4];
    const float* conv_w = (const float*)d_in[5];
    const float* conv_b = (const float*)d_in[6];
    const float* xproj_w= (const float*)d_in[7];
    const float* dt_w   = (const float*)d_in[8];
    const float* dt_b   = (const float*)d_in[9];
    const float* A_log  = (const float*)d_in[10];
    const float* D_vec  = (const float*)d_in[11];
    const float* out_w  = (const float*)d_in[12];
    const float* outp_w = (const float*)d_in[13];
    const float* bn_g   = (const float*)d_in[14];
    const float* bn_b   = (const float*)d_in[15];
    float* out = (float*)d_out;

    // ---- carve scratch (256B aligned slabs) ----
    char* wsp = (char*)d_ws;
    auto alloc = [&](size_t bytes) -> char* {
        char* p = wsp; wsp += (bytes + 255) & ~(size_t)255; return p;
    };
    float*    h     = (float*)   alloc((size_t)BL * DMODEL * 4);
    _Float16* hn    = (_Float16*)alloc((size_t)BL * DMODEL * 2);
    float*    xz    = (float*)   alloc((size_t)BL * 2 * DINNER * 4);
    float*    xcf[2]; _Float16* xch[2]; float* projb[2]; float* dtbuf[2]; float* ysb[2];
    for (int dir = 0; dir < 2; ++dir) {
        xcf[dir]   = (float*)   alloc((size_t)BL * DINNER * 4);
        xch[dir]   = (_Float16*)alloc((size_t)BL * DINNER * 2);
        projb[dir] = (float*)   alloc((size_t)BL * 48 * 4);
        dtbuf[dir] = (float*)   alloc((size_t)BL * DINNER * 4);
        ysb[dir]   = (float*)   alloc((size_t)BL * DINNER * 4);
    }
    _Float16* ycomb   = (_Float16*)alloc((size_t)BL * DINNER * 2);
    float*    tout    = (float*)   alloc((size_t)BL * FEATQ * 4);
    float*    stats   = (float*)   alloc((size_t)2 * FEATQ * 4);
    _Float16* inw_h   = (_Float16*)alloc((size_t)NLAYERS * 2 * DINNER * DMODEL * 2);
    _Float16* xpw_h   = (_Float16*)alloc((size_t)NLAYERS * 2 * 48 * DINNER * 2);
    _Float16* outw_h  = (_Float16*)alloc((size_t)NLAYERS * DMODEL * DINNER * 2);
    _Float16* outpw_h = (_Float16*)alloc((size_t)FEATQ * DMODEL * 2);

    // ---- weight conversions (f32 -> f16) ----
    auto cvt = [&](const float* s, _Float16* d, int n) {
        cvt_f32_f16_kernel<<<(n + 255) / 256, 256, 0, stream>>>(s, d, n);
    };
    cvt(in_w,    inw_h,   NLAYERS * 2 * DINNER * DMODEL);
    cvt(xproj_w, xpw_h,   NLAYERS * 2 * 48 * DINNER);
    cvt(out_w,   outw_h,  NLAYERS * DMODEL * DINNER);
    cvt(outp_w,  outpw_h, FEATQ * DMODEL);

    // ---- input projection ----
    input_proj_kernel<<<BL, 256, 0, stream>>>(x, inp_w, h);

    // ---- layer loop ----
    const int EPT = (BL * DINNER) / 256;   // element-wise grid
    for (int i = 0; i < NLAYERS; ++i) {
        layernorm_kernel<<<BL, 256, 0, stream>>>(h, ln_g + i * DMODEL, ln_b + i * DMODEL, hn);

        // in_proj: (BL,256) x (1024,256)^T -> xz (BL,1024)
        gemm_wmma_f16_kernel<4, 2 * DINNER, DMODEL, false>
            <<<dim3(BL / 64, (2 * DINNER) / 64), 128, 0, stream>>>(
                hn, inw_h + (size_t)i * 2 * DINNER * DMODEL, nullptr, xz);

        for (int dir = 0; dir < 2; ++dir) {
            int ld = i * 2 + dir;
            conv_silu_kernel<<<EPT, 256, 0, stream>>>(
                xz, conv_w + (size_t)ld * DINNER * DCONV, conv_b + (size_t)ld * DINNER,
                dir, xcf[dir], xch[dir]);

            // x_proj: (BL,512) x (48,512)^T -> proj (BL,48)
            gemm_wmma_f16_kernel<3, 48, DINNER, false>
                <<<dim3(BL / 64, 1), 128, 0, stream>>>(
                    xch[dir], xpw_h + (size_t)ld * 48 * DINNER, nullptr, projb[dir]);

            dt_kernel<<<EPT, 256, 0, stream>>>(
                projb[dir], dt_w + (size_t)ld * DINNER * DTRANK,
                dt_b + (size_t)ld * DINNER, dtbuf[dir]);

            scan_kernel<<<(BQ * DINNER) / 128, 128, 0, stream>>>(
                xcf[dir], dtbuf[dir], projb[dir],
                A_log + (size_t)ld * DINNER * DSTATE, D_vec + (size_t)ld * DINNER, ysb[dir]);
        }

        combine_kernel<<<EPT, 256, 0, stream>>>(xz, ysb[0], ysb[1], ycomb);

        // out_proj + residual into h: (BL,512) x (256,512)^T + h -> h
        gemm_wmma_f16_kernel<4, DMODEL, DINNER, true>
            <<<dim3(BL / 64, DMODEL / 64), 128, 0, stream>>>(
                ycomb, outw_h + (size_t)i * DMODEL * DINNER, h, h);
    }

    // ---- output projection + BatchNorm ----
    cvt(h, hn, BL * DMODEL);
    gemm_wmma_f16_kernel<4, FEATQ, DMODEL, false>
        <<<dim3(BL / 64, FEATQ / 64), 128, 0, stream>>>(hn, outpw_h, nullptr, tout);
    bnstats_kernel<<<FEATQ, 256, 0, stream>>>(tout, stats);
    bnapply_kernel<<<(BL * FEATQ) / 256, 256, 0, stream>>>(tout, stats, bn_g, bn_b, out);
}
